// TransformerEncoderLayer_86474871538004
// MI455X (gfx1250) — matrix operand order repaired and checked by reading, not compiled, not run on hardware
//
#include <hip/hip_runtime.h>
#include <hip/hip_bf16.h>
#include <cmath>

typedef __attribute__((ext_vector_type(16))) _Float16 v16h;
typedef __attribute__((ext_vector_type(8)))  float    v8f;
typedef __attribute__((ext_vector_type(4)))  unsigned int u32x4;
typedef __attribute__((ext_vector_type(8)))  int      i32x8;
typedef __attribute__((ext_vector_type(4)))  int      i32x4;

#define D_MODEL  768
#define D_POS    64
#define NHEAD    12
#define D_FF     3072
#define HEAD_DIM 64
#define SEQ_L    256
#define NBATCH   4
#define MTOT     (SEQ_L * NBATCH)   // 1024 token rows

#if defined(__has_builtin)
#if __has_builtin(__builtin_amdgcn_tensor_load_to_lds) && __has_builtin(__builtin_amdgcn_s_wait_tensorcnt)
#define HAS_TDM 1
#endif
#endif
#ifndef HAS_TDM
#define HAS_TDM 0
#endif

// ---------------------------------------------------------------- helpers

__device__ __forceinline__ float gelu_exact(float x) {
    return 0.5f * x * (1.0f + erff(x * 0.70710678118654752440f));
}

__device__ __forceinline__ v8f v8f_zero() {
    v8f z;
#pragma unroll
    for (int i = 0; i < 8; ++i) z[i] = 0.0f;
    return z;
}

// A fragment (16x32 f16, M x K). ISA 7.12.2: lanes 0-15 -> M=lane; halves 0-7
// hold K=0..7, halves 8-15 hold K=16..23; lanes 16-31 add +8 to K.
__device__ __forceinline__ v16h frag_a_lds(const _Float16* base, int ld, int lane) {
    const _Float16* p = base + (lane & 15) * ld + ((lane & 16) >> 1);
    v16h f;
#pragma unroll
    for (int h = 0; h < 16; ++h) f[h] = p[h + (h & 8)];
    return f;
}

// B fragment (32x16 f16, K x N). Lanes 0-15: col=lane; halves 0-7 hold K=0..7,
// halves 8-15 hold K=16..23; lanes 16-31 add +8 to K.
__device__ __forceinline__ v16h frag_b_lds(const _Float16* base, int ld, int lane) {
    const _Float16* p = base + (lane & 15) * ld + ((lane & 16) >> 1);
    v16h f;
#pragma unroll
    for (int h = 0; h < 16; ++h) f[h] = p[h + (h & 8)];
    return f;
}

#if HAS_TDM
// Issue a TDM 2-D tile load: rows x cols(halves) from global (row stride in
// halves) into contiguous LDS rows of `cols` halves. D# per CDNA5 ISA §8.3/8.4.
__device__ __forceinline__ void tdm_load_2d(const _Float16* gptr, unsigned lds_off,
                                            int rows, int cols, long row_stride) {
    const unsigned long long ga = (unsigned long long)(uintptr_t)gptr;
    u32x4 g0;
    g0[0] = 1u;                                             // count=1, user mode
    g0[1] = lds_off;                                        // lds_addr (bytes)
    g0[2] = (unsigned)(ga & 0xFFFFFFFFu);                   // global_addr lo
    g0[3] = (unsigned)((ga >> 32) & 0x01FFFFFFu) | 0x80000000u;  // addr hi | type=2
    const unsigned td0 = 0x40000000u, td1 = 0x40000000u;    // huge dims: no clip
    i32x8 g1;
    g1[0] = 0x00010000;                                     // data_size=1 (2 bytes)
    g1[1] = (int)((td0 & 0xFFFFu) << 16);                   // tensor_dim0 lo16
    g1[2] = (int)(((td0 >> 16) & 0xFFFFu) | ((td1 & 0xFFFFu) << 16));
    g1[3] = (int)(((td1 >> 16) & 0xFFFFu) | ((unsigned)cols << 16)); // tile_dim0
    g1[4] = rows;                                           // tile_dim1 (tile_dim2=0)
    g1[5] = (int)(row_stride & 0xFFFFFFFF);                 // tensor_dim0_stride lo
    g1[6] = (int)((row_stride >> 32) & 0xFFFF);             // stride hi16
    g1[7] = 0;
    i32x4 g2 = {0, 0, 0, 0};
    i32x4 g3 = {0, 0, 0, 0};
#if defined(__clang_major__) && __clang_major__ >= 23
    i32x8 g4 = {0, 0, 0, 0, 0, 0, 0, 0};
    __builtin_amdgcn_tensor_load_to_lds(g0, g1, g2, g3, g4, 0);
#else
    __builtin_amdgcn_tensor_load_to_lds(g0, g1, g2, g3, 0);
#endif
}
#endif

// ---------------------------------------------------------------- big batched GEMM
// 256 threads (8 waves), block tile 128x128, wave tile 32x64, K step 32.
// C[m,o] = scale*(sum_k A[m,k]*B[o,k] + bias[o]) (+resid) (+gelu)
// TDM double-buffered LDS feed when available; cooperative loads otherwise.
__global__ __launch_bounds__(256) void gemm_f16_big_kernel(
    const _Float16* __restrict__ A, int lda, long sA0, long sA1,
    const _Float16* __restrict__ B, int ldb, long sB0, long sB1,
    const float* __restrict__ bias,
    const float* __restrict__ resid, int ldr,
    float* __restrict__ Cf, _Float16* __restrict__ Ch, int ldc, long sC0, long sC1,
    int K, int Z0, int op, float scale)
{
    const int z  = blockIdx.z;
    const int z0 = z % Z0;
    const int z1 = z / Z0;
    A += (long)z0 * sA0 + (long)z1 * sA1;
    B += (long)z0 * sB0 + (long)z1 * sB1;
    const long cofs = (long)z0 * sC0 + (long)z1 * sC1;

    __shared__ alignas(16) _Float16 As[2][128][32];
    __shared__ alignas(16) _Float16 Bs[2][128][32];

    const int tid  = threadIdx.x;
    const int wave = tid >> 5;
    const int lane = tid & 31;
    const int wm   = wave & 3;     // 4 row-groups of 32
    const int wn   = wave >> 2;    // 2 col-groups of 64
    const int bm   = blockIdx.y * 128;
    const int bn   = blockIdx.x * 128;

    v8f acc[2][4];
#pragma unroll
    for (int s = 0; s < 2; ++s)
#pragma unroll
        for (int t = 0; t < 4; ++t) acc[s][t] = v8f_zero();

    const int nsteps = K >> 5;

#if HAS_TDM
    if (wave == 0) {
        tdm_load_2d(A + (long)bm * lda, (unsigned)(uintptr_t)&As[0][0][0], 128, 32, lda);
        tdm_load_2d(B + (long)bn * ldb, (unsigned)(uintptr_t)&Bs[0][0][0], 128, 32, ldb);
    }
#else
    const int lrow  = tid & 127;
    const int lpart = (tid >> 7) << 4;   // 0 or 16 halves
#endif

    for (int s = 0; s < nsteps; ++s) {
#if HAS_TDM
        const int buf = s & 1;
        __syncthreads();   // everyone done with buf^1 before DMA overwrites it
        if (s + 1 < nsteps) {
            if (wave == 0) {
                const int k1 = (s + 1) << 5;
                tdm_load_2d(A + (long)bm * lda + k1,
                            (unsigned)(uintptr_t)&As[buf ^ 1][0][0], 128, 32, lda);
                tdm_load_2d(B + (long)bn * ldb + k1,
                            (unsigned)(uintptr_t)&Bs[buf ^ 1][0][0], 128, 32, ldb);
            }
            __builtin_amdgcn_s_wait_tensorcnt(2);  // loads for buf have landed
        } else {
            __builtin_amdgcn_s_wait_tensorcnt(0);
        }
        __syncthreads();   // publish buf to all waves
#else
        const int buf = 0;
        const int k0  = s << 5;
        __syncthreads();
        {
            const uint4* pa = (const uint4*)(A + (long)(bm + lrow) * lda + k0 + lpart);
            uint4 a0 = pa[0], a1 = pa[1];
            *(uint4*)&As[0][lrow][lpart]     = a0;
            *(uint4*)&As[0][lrow][lpart + 8] = a1;
            const uint4* pb = (const uint4*)(B + (long)(bn + lrow) * ldb + k0 + lpart);
            uint4 b0 = pb[0], b1 = pb[1];
            *(uint4*)&Bs[0][lrow][lpart]     = b0;
            *(uint4*)&Bs[0][lrow][lpart + 8] = b1;
        }
        __syncthreads();
#endif
        const v16h af0 = frag_a_lds(&As[buf][wm * 32][0], 32, lane);
        const v16h af1 = frag_a_lds(&As[buf][wm * 32 + 16][0], 32, lane);
#pragma unroll
        for (int t = 0; t < 4; ++t) {
            v16h bf = frag_b_lds(&Bs[buf][wn * 64 + t * 16][0], 32, lane);
            acc[0][t] = __builtin_amdgcn_wmma_f32_16x16x32_f16(
                false, af0, false, bf, (short)0, acc[0][t], false, false);
            acc[1][t] = __builtin_amdgcn_wmma_f32_16x16x32_f16(
                false, af1, false, bf, (short)0, acc[1][t], false, false);
        }
    }

    const int col = lane & 15;
    const int rhi = (lane & 16) ? 8 : 0;
#pragma unroll
    for (int t = 0; t < 4; ++t) {
        const int o  = bn + wn * 64 + t * 16 + col;
        const float bv = bias ? bias[o] : 0.0f;
#pragma unroll
        for (int sub = 0; sub < 2; ++sub) {
#pragma unroll
            for (int r = 0; r < 8; ++r) {
                const int m = bm + wm * 32 + sub * 16 + rhi + r;
                float v = (acc[sub][t][r] + bv) * scale;
                if (resid) v += resid[(long)m * ldr + o];
                if (op == 1) v = gelu_exact(v);
                const long idx = cofs + (long)m * ldc + o;
                if (Cf) Cf[idx] = v;
                if (Ch) Ch[idx] = (_Float16)v;
            }
        }
    }
}

// ---------------------------------------------------------------- small batched GEMM
// 128 threads (4 waves), block tile 64x64 (for Nout=64 cases: qp, attn@V).
__global__ __launch_bounds__(128) void gemm_f16_kernel(
    const _Float16* __restrict__ A, int lda, long sA0, long sA1,
    const _Float16* __restrict__ B, int ldb, long sB0, long sB1,
    const float* __restrict__ bias,
    const float* __restrict__ resid, int ldr,
    float* __restrict__ Cf, _Float16* __restrict__ Ch, int ldc, long sC0, long sC1,
    int K, int Z0, int op, float scale)
{
    const int z  = blockIdx.z;
    const int z0 = z % Z0;
    const int z1 = z / Z0;
    A += (long)z0 * sA0 + (long)z1 * sA1;
    B += (long)z0 * sB0 + (long)z1 * sB1;
    const long cofs = (long)z0 * sC0 + (long)z1 * sC1;

    __shared__ alignas(16) _Float16 As[64][40];
    __shared__ alignas(16) _Float16 Bs[64][40];

    const int tid  = threadIdx.x;
    const int wave = tid >> 5;
    const int lane = tid & 31;
    const int bm   = blockIdx.y * 64;
    const int bn   = blockIdx.x * 64;

    v8f acc[4];
#pragma unroll
    for (int t = 0; t < 4; ++t) acc[t] = v8f_zero();

    const int lrow  = tid & 63;
    const int lpart = (tid >> 6) << 4;   // 0 or 16 halves

    for (int k0 = 0; k0 < K; k0 += 32) {
        __syncthreads();
        {
            const uint4* pa = (const uint4*)(A + (long)(bm + lrow) * lda + k0 + lpart);
            uint4 a0 = pa[0], a1 = pa[1];
            *(uint4*)&As[lrow][lpart]     = a0;
            *(uint4*)&As[lrow][lpart + 8] = a1;
            const uint4* pb = (const uint4*)(B + (long)(bn + lrow) * ldb + k0 + lpart);
            uint4 b0 = pb[0], b1 = pb[1];
            *(uint4*)&Bs[lrow][lpart]     = b0;
            *(uint4*)&Bs[lrow][lpart + 8] = b1;
        }
        __syncthreads();
        v16h af = frag_a_lds(&As[wave * 16][0], 40, lane);
#pragma unroll
        for (int t = 0; t < 4; ++t) {
            v16h bf = frag_b_lds(&Bs[t * 16][0], 40, lane);
            acc[t] = __builtin_amdgcn_wmma_f32_16x16x32_f16(
                false, af, false, bf, (short)0, acc[t], false, false);
        }
    }

    const int col = lane & 15;
    const int rhi = (lane & 16) ? 8 : 0;
#pragma unroll
    for (int t = 0; t < 4; ++t) {
        const int o  = bn + t * 16 + col;
        const float bv = bias ? bias[o] : 0.0f;
#pragma unroll
        for (int r = 0; r < 8; ++r) {
            const int m = bm + wave * 16 + rhi + r;
            float v = (acc[t][r] + bv) * scale;
            if (resid) v += resid[(long)m * ldr + o];
            if (op == 1) v = gelu_exact(v);
            const long idx = cofs + (long)m * ldc + o;
            if (Cf) Cf[idx] = v;
            if (Ch) Ch[idx] = (_Float16)v;
        }
    }
}

// ---------------------------------------------------------------- content-position scores
// scores[n,h,i,j] += sum_c qp[h][i*4+n][c] * pos[n,i,j,c]   (h padded 12->16)
__global__ __launch_bounds__(128) void pos_score_kernel(
    const _Float16* __restrict__ qp,   // [NHEAD][MTOT][64]
    const float* __restrict__ pos,     // (N,L,L,64) f32
    float* __restrict__ scores)        // (N,H,L,L) f32
{
    const int b = blockIdx.x;          // n*L + i
    const int n = b / SEQ_L;
    const int i = b % SEQ_L;
    const int m = i * NBATCH + n;

    __shared__ alignas(16) _Float16 aq[16][72];
    __shared__ alignas(16) _Float16 bs[64][72];

    const int tid  = threadIdx.x;
    const int wave = tid >> 5;
    const int lane = tid & 31;

    {   // load qp rows h=0..15 (zero-pad h>=12): 16 rows x 64 halves
        const int row = tid >> 3;
        const int seg = (tid & 7) << 3;
        uint4 val = make_uint4(0u, 0u, 0u, 0u);
        if (row < NHEAD)
            val = *(const uint4*)(qp + (long)row * (MTOT * 64) + (long)m * 64 + seg);
        *(uint4*)&aq[row][seg] = val;
    }
    __syncthreads();
    const v16h af0 = frag_a_lds(&aq[0][0], 72, lane);
    const v16h af1 = frag_a_lds(&aq[0][32], 72, lane);

    const int col = lane & 15;
    const int rhi = (lane & 16) ? 8 : 0;

    for (int jc = 0; jc < 4; ++jc) {
        const int j0 = jc * 64;
        __syncthreads();
        {   // convert pos tile (64 j-rows x 64 c) f32 -> f16 in LDS
            const int row = tid >> 1;
            const int off = (tid & 1) * 32;
            const float4* src = (const float4*)(
                pos + ((long)(n * SEQ_L + i) * SEQ_L + (j0 + row)) * 64 + off);
#pragma unroll
            for (int q8 = 0; q8 < 8; ++q8) {
                float4 f = src[q8];
                const int c = off + q8 * 4;
                bs[row][c + 0] = (_Float16)f.x;
                bs[row][c + 1] = (_Float16)f.y;
                bs[row][c + 2] = (_Float16)f.z;
                bs[row][c + 3] = (_Float16)f.w;
            }
        }
        __syncthreads();
        v8f acc = v8f_zero();
        v16h bf0 = frag_b_lds(&bs[wave * 16][0], 72, lane);
        v16h bf1 = frag_b_lds(&bs[wave * 16][32], 72, lane);
        acc = __builtin_amdgcn_wmma_f32_16x16x32_f16(false, af0, false, bf0, (short)0, acc, false, false);
        acc = __builtin_amdgcn_wmma_f32_16x16x32_f16(false, af1, false, bf1, (short)0, acc, false, false);

        const int j = j0 + wave * 16 + col;
#pragma unroll
        for (int r = 0; r < 8; ++r) {
            const int h = rhi + r;
            if (h < NHEAD) {
                const long idx = ((long)(n * NHEAD + h) * SEQ_L + i) * SEQ_L + j;
                scores[idx] += acc[r];
            }
        }
    }
}

// ---------------------------------------------------------------- softmax over last dim (256)
__global__ __launch_bounds__(256) void softmax_kernel(
    const float* __restrict__ s, _Float16* __restrict__ attn)
{
    const long row = blockIdx.x;       // (n*H+h)*L + i
    const int  t   = threadIdx.x;
    __shared__ float red[256];
    const float v = s[row * SEQ_L + t];
    red[t] = v;
    __syncthreads();
    for (int st = 128; st > 0; st >>= 1) {
        if (t < st) red[t] = fmaxf(red[t], red[t + st]);
        __syncthreads();
    }
    const float mx = red[0];
    __syncthreads();
    const float e = expf(v - mx);
    red[t] = e;
    __syncthreads();
    for (int st = 128; st > 0; st >>= 1) {
        if (t < st) red[t] += red[t + st];
        __syncthreads();
    }
    attn[row * SEQ_L + t] = (_Float16)(e / red[0]);
}

// ---------------------------------------------------------------- LayerNorm over 768
__global__ __launch_bounds__(256) void layernorm_kernel(
    const float* __restrict__ x, const float* __restrict__ g, const float* __restrict__ bta,
    float* __restrict__ outf, _Float16* __restrict__ outh)
{
    const long mrow = blockIdx.x;
    const int  t    = threadIdx.x;
    const float* r  = x + mrow * D_MODEL;
    __shared__ float red[256];
    const float a0 = r[t], a1 = r[t + 256], a2 = r[t + 512];
    red[t] = a0 + a1 + a2;
    __syncthreads();
    for (int st = 128; st > 0; st >>= 1) {
        if (t < st) red[t] += red[t + st];
        __syncthreads();
    }
    const float mu = red[0] * (1.0f / 768.0f);
    __syncthreads();
    const float d0 = a0 - mu, d1 = a1 - mu, d2 = a2 - mu;
    red[t] = d0 * d0 + d1 * d1 + d2 * d2;
    __syncthreads();
    for (int st = 128; st > 0; st >>= 1) {
        if (t < st) red[t] += red[t + st];
        __syncthreads();
    }
    const float rstd = rsqrtf(red[0] * (1.0f / 768.0f) + 1e-5f);
    const float ds[3] = {d0, d1, d2};
#pragma unroll
    for (int kb = 0; kb < 3; ++kb) {
        const int e = t + kb * 256;
        const float y = ds[kb] * rstd * g[e] + bta[e];
        outf[mrow * D_MODEL + e] = y;
        if (outh) outh[mrow * D_MODEL + e] = (_Float16)y;
    }
}

// ---------------------------------------------------------------- small utility kernels
__global__ void cvt_kernel(const float* __restrict__ in, _Float16* __restrict__ out, int n) {
    const int i = blockIdx.x * 256 + threadIdx.x;
    if (i < n) out[i] = (_Float16)in[i];
}

// p_w (768,64) -> pwt (64,768) f16
__global__ void pwt_kernel(const float* __restrict__ pw, _Float16* __restrict__ pwt) {
    const int i = blockIdx.x * 256 + threadIdx.x;
    if (i < D_MODEL * D_POS) {
        const int o = i / D_POS, c = i % D_POS;
        pwt[c * D_MODEL + o] = (_Float16)pw[i];
    }
}

// v_h (m=j*4+n, e=h*64+d) -> vt[((n*H+h)*64+d)*256 + j]
__global__ void vt_kernel(const _Float16* __restrict__ vh, _Float16* __restrict__ vt) {
    const unsigned i = blockIdx.x * 256 + threadIdx.x;   // < 786432
    const int j = i & 255;
    unsigned rest = i >> 8;
    const int d = rest & 63;  rest >>= 6;
    const int h = rest % NHEAD;
    const int n = rest / NHEAD;
    vt[i] = vh[((long)j * NBATCH + n) * D_MODEL + h * HEAD_DIM + d];
}

// ---------------------------------------------------------------- host launcher
extern "C" void kernel_launch(void* const* d_in, const int* in_sizes, int n_in,
                              void* d_out, int out_size, void* d_ws, size_t ws_size,
                              hipStream_t stream)
{
    (void)in_sizes; (void)n_in; (void)out_size; (void)ws_size;
    const float* src    = (const float*)d_in[0];
    const float* pos    = (const float*)d_in[1];
    const float* q_w    = (const float*)d_in[2];
    const float* q_b    = (const float*)d_in[3];
    const float* k_w    = (const float*)d_in[4];
    const float* k_b    = (const float*)d_in[5];
    const float* v_w    = (const float*)d_in[6];
    const float* v_b    = (const float*)d_in[7];
    const float* p_w    = (const float*)d_in[8];
    const float* out_w  = (const float*)d_in[10];
    const float* out_b  = (const float*)d_in[11];
    const float* lin1_w = (const float*)d_in[12];
    const float* lin1_b = (const float*)d_in[13];
    const float* lin2_w = (const float*)d_in[14];
    const float* lin2_b = (const float*)d_in[15];
    const float* n1_g   = (const float*)d_in[16];
    const float* n1_b   = (const float*)d_in[17];
    const float* n2_g   = (const float*)d_in[18];
    const float* n2_b   = (const float*)d_in[19];

    char* ws = (char*)d_ws;
    size_t off = 0;
    auto take = [&](size_t bytes) -> char* {
        char* p = ws + off;
        off += (bytes + 255) & ~(size_t)255;
        return p;
    };

    const size_t ME  = (size_t)MTOT * D_MODEL;           // 786432
    _Float16* src_h = (_Float16*)take(ME * 2);
    _Float16* Wq    = (_Float16*)take((size_t)D_MODEL * D_MODEL * 2);
    _Float16* Wk    = (_Float16*)take((size_t)D_MODEL * D_MODEL * 2);
    _Float16* Wv    = (_Float16*)take((size_t)D_MODEL * D_MODEL * 2);
    _Float16* Wo    = (_Float16*)take((size_t)D_MODEL * D_MODEL * 2);
    _Float16* W1    = (_Float16*)take((size_t)D_FF * D_MODEL * 2);
    _Float16* W2    = (_Float16*)take((size_t)D_MODEL * D_FF * 2);
    _Float16* pwt   = (_Float16*)take((size_t)D_POS * D_MODEL * 2);
    _Float16* q_h   = (_Float16*)take(ME * 2);
    _Float16* k_h   = (_Float16*)take(ME * 2);
    _Float16* v_h   = (_Float16*)take(ME * 2);
    _Float16* vt    = (_Float16*)take((size_t)NBATCH * NHEAD * HEAD_DIM * SEQ_L * 2);
    _Float16* qp    = (_Float16*)take((size_t)NHEAD * MTOT * 64 * 2);
    float*    scores= (float*)   take((size_t)NBATCH * NHEAD * SEQ_L * SEQ_L * 4);
    _Float16* attn  = (_Float16*)take((size_t)NBATCH * NHEAD * SEQ_L * SEQ_L * 2);
    _Float16* ctx_h = (_Float16*)take(ME * 2);
    float*    x1    = (float*)   take(ME * 4);
    float*    x1n   = (float*)   take(ME * 4);
    _Float16* x1n_h = (_Float16*)take(ME * 2);
    _Float16* g_h   = (_Float16*)take((size_t)MTOT * D_FF * 2);
    float*    y     = (float*)   take(ME * 4);

    auto gemm = [&](const _Float16* A, int lda, long sA0, long sA1,
                    const _Float16* B, int ldb, long sB0, long sB1,
                    const float* bias, const float* resid, int ldr,
                    float* Cf, _Float16* Ch, int ldc, long sC0, long sC1,
                    int M, int Nout, int K, int Z, int Z0, int op, float scale) {
        if ((M % 128 == 0) && (Nout % 128 == 0)) {
            dim3 grid(Nout / 128, M / 128, Z);
            gemm_f16_big_kernel<<<grid, dim3(256), 0, stream>>>(
                A, lda, sA0, sA1, B, ldb, sB0, sB1, bias, resid, ldr,
                Cf, Ch, ldc, sC0, sC1, K, Z0, op, scale);
        } else {
            dim3 grid(Nout / 64, M / 64, Z);
            gemm_f16_kernel<<<grid, dim3(128), 0, stream>>>(
                A, lda, sA0, sA1, B, ldb, sB0, sB1, bias, resid, ldr,
                Cf, Ch, ldc, sC0, sC1, K, Z0, op, scale);
        }
    };

    // ---- f32 -> f16 staging
    cvt_kernel<<<(int)((ME + 255) / 256), 256, 0, stream>>>(src, src_h, (int)ME);
    cvt_kernel<<<(768 * 768 + 255) / 256, 256, 0, stream>>>(q_w,   Wq, 768 * 768);
    cvt_kernel<<<(768 * 768 + 255) / 256, 256, 0, stream>>>(k_w,   Wk, 768 * 768);
    cvt_kernel<<<(768 * 768 + 255) / 256, 256, 0, stream>>>(v_w,   Wv, 768 * 768);
    cvt_kernel<<<(768 * 768 + 255) / 256, 256, 0, stream>>>(out_w, Wo, 768 * 768);
    cvt_kernel<<<(D_FF * 768 + 255) / 256, 256, 0, stream>>>(lin1_w, W1, D_FF * 768);
    cvt_kernel<<<(768 * D_FF + 255) / 256, 256, 0, stream>>>(lin2_w, W2, 768 * D_FF);
    pwt_kernel<<<(D_MODEL * D_POS + 255) / 256, 256, 0, stream>>>(p_w, pwt);

    // ---- Q/K/V projections (q pre-scaled by hd^-0.5)
    gemm(src_h, 768, 0, 0, Wq, 768, 0, 0, q_b, nullptr, 0,
         nullptr, q_h, 768, 0, 0, MTOT, 768, 768, 1, 1, 0, 0.125f);
    gemm(src_h, 768, 0, 0, Wk, 768, 0, 0, k_b, nullptr, 0,
         nullptr, k_h, 768, 0, 0, MTOT, 768, 768, 1, 1, 0, 1.0f);
    gemm(src_h, 768, 0, 0, Wv, 768, 0, 0, v_b, nullptr, 0,
         nullptr, v_h, 768, 0, 0, MTOT, 768, 768, 1, 1, 0, 1.0f);
    vt_kernel<<<(int)(ME / 256), 256, 0, stream>>>(v_h, vt);

    // ---- qp[h][m][c] = q_h[m, h*64+:64] @ p_w[h*64+:64, :]  (batched over h)
    gemm(q_h, 768, 64, 0, pwt, 768, 64, 0, nullptr, nullptr, 0,
         nullptr, qp, 64, (long)MTOT * 64, 0,
         MTOT, 64, 64, NHEAD, NHEAD, 0, 1.0f);

    // ---- content-content scores: q @ k^T per (n,h)
    gemm(q_h, NBATCH * 768, 64, 768, k_h, NBATCH * 768, 64, 768, nullptr, nullptr, 0,
         scores, nullptr, SEQ_L, (long)SEQ_L * SEQ_L, (long)NHEAD * SEQ_L * SEQ_L,
         SEQ_L, SEQ_L, 64, NBATCH * NHEAD, NHEAD, 0, 1.0f);

    // ---- content-position scores (adds into scores)
    pos_score_kernel<<<NBATCH * SEQ_L, 128, 0, stream>>>(qp, pos, scores);

    // ---- softmax -> f16 attention
    softmax_kernel<<<NBATCH * NHEAD * SEQ_L, 256, 0, stream>>>(scores, attn);

    // ---- ctx = attn @ v per (n,h), scattered to (L,N,E) f16
    gemm(attn, SEQ_L, (long)SEQ_L * SEQ_L, (long)NHEAD * SEQ_L * SEQ_L,
         vt, SEQ_L, (long)HEAD_DIM * SEQ_L, (long)NHEAD * HEAD_DIM * SEQ_L,
         nullptr, nullptr, 0,
         nullptr, ctx_h, NBATCH * 768, 64, 768,
         SEQ_L, HEAD_DIM, SEQ_L, NBATCH * NHEAD, NHEAD, 0, 1.0f);

    // ---- out projection + residual(src), then LN1
    gemm(ctx_h, 768, 0, 0, Wo, 768, 0, 0, out_b, src, 768,
         x1, nullptr, 768, 0, 0, MTOT, 768, 768, 1, 1, 0, 1.0f);
    layernorm_kernel<<<MTOT, 256, 0, stream>>>(x1, n1_g, n1_b, x1n, x1n_h);

    // ---- FFN: gelu(x1n @ W1^T + b1) @ W2^T + b2 + x1n, then LN2 -> out
    gemm(x1n_h, 768, 0, 0, W1, 768, 0, 0, lin1_b, nullptr, 0,
         nullptr, g_h, D_FF, 0, 0, MTOT, D_FF, 768, 1, 1, 1, 1.0f);
    gemm(g_h, D_FF, 0, 0, W2, D_FF, 0, 0, lin2_b, x1n, 768,
         y, nullptr, 768, 0, 0, MTOT, 768, D_FF, 1, 1, 0, 1.0f);
    layernorm_kernel<<<MTOT, 256, 0, stream>>>(y, n2_g, n2_b, (float*)d_out, nullptr);
}
